// WindowedCausalSelfAttention_65627100282993
// MI455X (gfx1250) — matrix operand order repaired and checked by reading, not compiled, not run on hardware
//
#include <hip/hip_runtime.h>

typedef __attribute__((ext_vector_type(16))) _Float16 v16h;
typedef __attribute__((ext_vector_type(8)))  _Float16 v8h;
typedef __attribute__((ext_vector_type(8)))  float    v8f;
typedef __attribute__((ext_vector_type(4)))  float    v4f;
typedef __attribute__((ext_vector_type(4)))  unsigned int u32x4;
typedef __attribute__((ext_vector_type(8)))  int      i32x8;
typedef __attribute__((ext_vector_type(4)))  int      i32x4;

#define WMMA_F16(A,B,C) __builtin_amdgcn_wmma_f32_16x16x32_f16(false,(A),false,(B),(short)0,(C),false,false)

#if __has_builtin(__builtin_amdgcn_tensor_load_to_lds) && __has_builtin(__builtin_amdgcn_s_wait_tensorcnt)
#define USE_TDM 1
#else
#define USE_TDM 0
#endif

static constexpr int kB  = 4;
static constexpr int kS  = 4096;   // P == S (4096 % 256 == 0, no pad)
static constexpr int kD  = 1024;
static constexpr int kH  = 16;
static constexpr int kHd = 64;
static constexpr int kW  = 256;
static constexpr float kEps = 1.1920928955078125e-07f;

__device__ __forceinline__ v16h cat8(v8h lo, v8h hi){
  v16h r;
#pragma unroll
  for (int i=0;i<8;i++){ r[i]=lo[i]; r[8+i]=hi[i]; }
  return r;
}
__device__ __forceinline__ v8f vzero8(){
  v8f z;
#pragma unroll
  for (int i=0;i<8;i++) z[i]=0.f;
  return z;
}

#if USE_TDM
// ---------------------------------------------------------------------------
// Tensor Data Mover: 2D f16 tile (tile_x elems/row, tile_rows rows, row stride
// stride_elems) global -> LDS, with LDS row padding (pad_int/pad_amt codes per
// ISA D# group1: interval 0=2dw..7=256dw, amount 0=1dw..127=128dw).
// D# group0: count=1, lds_addr, global_addr, type=2. Remaining groups zero
// (2D tile, no gather/iterate/multicast). 6-arg builtin on this toolchain.
// Low 32 bits of a generic __shared__ pointer == LDS byte offset (aperture map).
// ---------------------------------------------------------------------------
__device__ __forceinline__ void tdm_load_2d(const void* lds, const void* gptr,
                                            unsigned tile_x, unsigned tile_rows,
                                            unsigned stride_elems,
                                            unsigned pad_int, unsigned pad_amt)
{
  unsigned long long ga = (unsigned long long)(size_t)gptr;
  unsigned la = (unsigned)(size_t)lds;
  u32x4 g0;
  g0[0] = 1u;                                          // count=1, user mode
  g0[1] = la;                                          // lds_addr
  g0[2] = (unsigned)(ga & 0xFFFFFFFFu);                // global_addr[31:0]
  g0[3] = (unsigned)((ga >> 32) & 0x01FFFFFFu) | (2u << 30);  // [56:32] | type=2
  i32x8 g1;
  g1[0] = (int)((1u<<16) | (1u<<20) | (pad_int<<22) | (pad_amt<<25)); // f16, pad
  const unsigned td0 = 1u<<30, td1 = 1u<<30;           // huge dims: never clip
  g1[1] = (int)((td0 & 0xFFFFu) << 16);
  g1[2] = (int)((td0 >> 16) | ((td1 & 0xFFFFu) << 16));
  g1[3] = (int)((td1 >> 16) | (tile_x << 16));
  g1[4] = (int)(tile_rows & 0xFFFFu);                  // tile_dim1 (tile_dim2=0)
  g1[5] = (int)stride_elems;                           // dim0_stride[31:0]
  g1[6] = 0;
  g1[7] = 0;
  i32x4 gz4;
  gz4[0]=0; gz4[1]=0; gz4[2]=0; gz4[3]=0;
  i32x8 gz8;
#pragma unroll
  for (int i=0;i<8;i++) gz8[i]=0;
  __builtin_amdgcn_tensor_load_to_lds(g0, g1, gz4, gz4, gz8, 0);
}
#endif

// ---------------------------------------------------------------------------
// Elementwise f32 -> f16 conversion (8 elems/thread)
// ---------------------------------------------------------------------------
__global__ __launch_bounds__(256)
void cvt_kernel(const float* __restrict__ src, _Float16* __restrict__ dst)
{
  size_t i = ((size_t)blockIdx.x * 256 + threadIdx.x) * 8;
  v4f a = *(const v4f*)(src + i);
  v4f b = *(const v4f*)(src + i + 4);
  v8h o;
#pragma unroll
  for (int j=0;j<4;j++){ o[j] = (_Float16)a[j]; o[4+j] = (_Float16)b[j]; }
  *(v8h*)(dst + i) = o;
}

// ---------------------------------------------------------------------------
// WMMA GEMM: C[M=16384,N=1024] = A[M,1024](f16) @ W[N,1024](f16)^T
// BM=128,BN=128,BK=32; 8 waves, wave tile 32x64 (2x4 C tiles).
// Staging via TDM double buffering (wave0 issues, TENSORcnt-gated) or
// synchronous f16 copies when TDM builtins are unavailable.
// MODE 0: epilogue scatters f16 into [B][H][P][64], weight = Wbase+z*1M.
// MODE 1: epilogue writes f32 row-major (d_out).
// ---------------------------------------------------------------------------
template<int MODE>
__global__ __launch_bounds__(256)
void gemm_kernel(const _Float16* __restrict__ Ap, const _Float16* __restrict__ Wbase,
                 _Float16* __restrict__ q16, _Float16* __restrict__ k16,
                 _Float16* __restrict__ v16, float* __restrict__ outp)
{
  __shared__ _Float16 sA[2][128*40];
  __shared__ _Float16 sB[2][128*40];
  const int tid  = threadIdx.x;
  const int wave = tid >> 5, lane = tid & 31;
  const int l16  = lane & 15, hf = lane >> 4;
  const int wm   = wave & 3, wn = wave >> 2;
  const int m0   = blockIdx.y * 128, n0 = blockIdx.x * 128;
  const _Float16* Wt = Wbase + (size_t)blockIdx.z * kD * kD;

  v8f acc[2][4];
#pragma unroll
  for (int i=0;i<2;i++)
#pragma unroll
    for (int j=0;j<4;j++) acc[i][j] = vzero8();

#if USE_TDM
  const bool issuer = (__builtin_amdgcn_readfirstlane(tid >> 5) == 0);
  if (issuer){
    tdm_load_2d(&sA[0][0], Ap + (size_t)m0*kD, 32, 128, kD, 3, 3);
    tdm_load_2d(&sB[0][0], Wt + (size_t)n0*kD, 32, 128, kD, 3, 3);
    __builtin_amdgcn_s_wait_tensorcnt((short)0);
  }
  __syncthreads();
#endif

  for (int it = 0; it < 32; ++it){
    const int buf = it & 1;
    const int kk  = it * 32;
#if USE_TDM
    if (issuer && it + 1 < 32){
      tdm_load_2d(&sA[buf^1][0], Ap + (size_t)m0*kD + kk + 32, 32, 128, kD, 3, 3);
      tdm_load_2d(&sB[buf^1][0], Wt + (size_t)n0*kD + kk + 32, 32, 128, kD, 3, 3);
    }
#else
    __syncthreads();
#pragma unroll
    for (int q=0;q<2;q++){
      int e = tid*2 + q;                 // 0..511
      int r = e >> 2, c = e & 3;
      *(v8h*)&sA[buf][r*40 + c*8] = *(const v8h*)(Ap + (size_t)(m0+r)*kD + kk + c*8);
      *(v8h*)&sB[buf][r*40 + c*8] = *(const v8h*)(Wt + (size_t)(n0+r)*kD + kk + c*8);
    }
    __syncthreads();
#endif

    v16h a[2];
    const int ab = hf ? 16 : 0;
#pragma unroll
    for (int mt=0; mt<2; mt++){
      int r = wm*32 + mt*16 + l16;
      a[mt] = cat8(*(const v8h*)&sA[buf][r*40 + ab],
                   *(const v8h*)&sA[buf][r*40 + 8 + ab]);
    }
#pragma unroll
    for (int nt=0; nt<4; nt++){
      int cc = wn*64 + nt*16 + l16;
      int kb2 = hf ? 16 : 0;
      v16h bop = cat8(*(const v8h*)&sB[buf][cc*40 + kb2],
                      *(const v8h*)&sB[buf][cc*40 + kb2 + 8]);
#pragma unroll
      for (int mt=0; mt<2; mt++)
        acc[mt][nt] = WMMA_F16(a[mt], bop, acc[mt][nt]);
    }

#if USE_TDM
    if (issuer && it + 1 < 32) __builtin_amdgcn_s_wait_tensorcnt((short)0);
    __syncthreads();
#endif
  }

  // Epilogue. C layout: VGPR j holds (M = j + 8*hf, N = l16).
#pragma unroll
  for (int mt=0; mt<2; mt++)
#pragma unroll
  for (int nt=0; nt<4; nt++)
#pragma unroll
  for (int j=0; j<8; j++){
    int gm = m0 + wm*32 + mt*16 + j + 8*hf;
    int gn = n0 + wn*64 + nt*16 + l16;
    if (MODE == 0){
      int bb = gm >> 12;
      int p  = gm & 4095;
      int h  = gn >> 6;
      int hd = gn & 63;
      _Float16* dst = (blockIdx.z==0) ? q16 : (blockIdx.z==1 ? k16 : v16);
      dst[((((size_t)bb*kH + h)*kS + p) << 6) + hd] = (_Float16)acc[mt][nt][j];
    } else {
      outp[(size_t)gm*kD + gn] = acc[mt][nt][j];
    }
  }
}

// ---------------------------------------------------------------------------
// In-place RMS norm + RoPE (+ q_gain for Q) on f16 [B][H][P][64] buffers.
// ---------------------------------------------------------------------------
__global__ __launch_bounds__(256)
void rmsrope_kernel(_Float16* __restrict__ q16, _Float16* __restrict__ k16,
                    const float* __restrict__ qgain)
{
  const int row  = blockIdx.x * blockDim.x + threadIdx.x;  // [b][h][p] flat
  const bool isQ = (blockIdx.y == 0);
  _Float16* base = (isQ ? q16 : k16) + (size_t)row * kHd;
  const int p = row & (kS - 1);
  const int h = (row >> 12) & (kH - 1);

  float x[64];
#pragma unroll
  for (int c=0;c<8;c++){
    v8h v = *(const v8h*)(base + c*8);
#pragma unroll
    for (int i=0;i<8;i++) x[c*8+i] = (float)v[i];
  }
  float ss = 0.f;
#pragma unroll
  for (int i=0;i<64;i++) ss += x[i]*x[i];
  const float r = rsqrtf(ss * (1.f/64.f) + kEps);
  const float g = isQ ? qgain[h] : 1.f;

  float o[64];
#pragma unroll
  for (int j=0;j<32;j++){
    float inv = __expf(-(float)j * 0.28782313662425576f); // ln(10000)/32
    float ang = (float)p * inv;
    float sn, cs;
    __sincosf(ang, &sn, &cs);
    float a = x[j]*r, b2 = x[j+32]*r;
    o[j]    = ( a*cs + b2*sn) * g;
    o[j+32] = (-a*sn + b2*cs) * g;
  }
#pragma unroll
  for (int c=0;c<8;c++){
    v8h v;
#pragma unroll
    for (int i=0;i<8;i++) v[i] = (_Float16)o[c*8+i];
    *(v8h*)(base + c*8) = v;
  }
}

// ---------------------------------------------------------------------------
// Windowed causal attention. One workgroup per (head, window, batch).
// 16 waves; wave `qt` owns q-rows [qt*16, qt*16+16). Flash softmax over
// key blocks of 64. K staged via TDM (padded [key][hd]), V transposed in LDS.
// ---------------------------------------------------------------------------
__global__ __launch_bounds__(512)
void attn_kernel(const _Float16* __restrict__ q16, const _Float16* __restrict__ k16,
                 const _Float16* __restrict__ v16, _Float16* __restrict__ a16)
{
  __shared__ _Float16 Ks[64*72];       // [key][hd], row stride 72 halves
  __shared__ _Float16 Vt[64*72];       // [hd][key], row stride 72 halves
  __shared__ _Float16 Ps[16][16*40];   // per-wave P staging, 16x32 tile
  const int h = blockIdx.x, wdw = blockIdx.y, b = blockIdx.z;
  const int tid  = threadIdx.x;
  const int wave = tid >> 5, lane = tid & 31;
  const int l16  = lane & 15, hf = lane >> 4;
  const int qt   = wave;               // q-tile index 0..15
  const size_t headoff = ((((size_t)b*kH + h)*kS) + (size_t)wdw*kW) * kHd;
  const _Float16* Qp = q16 + headoff;
  const _Float16* Kp = k16 + headoff;
  const _Float16* Vp = v16 + headoff;
#if USE_TDM
  const bool issuer = (__builtin_amdgcn_readfirstlane(tid >> 5) == 0);
#endif

  const int ab = hf ? 16 : 0;
  v16h Aq[2];
#pragma unroll
  for (int t=0;t<2;t++){
    const _Float16* qr = Qp + (size_t)(qt*16 + l16)*kHd + t*32;
    Aq[t] = cat8(*(const v8h*)(qr + ab), *(const v8h*)(qr + 8 + ab));
  }

  v8f acc[4];
#pragma unroll
  for (int i=0;i<4;i++) acc[i] = vzero8();
  float m_arr[8], l_arr[8];
#pragma unroll
  for (int j=0;j<8;j++){ m_arr[j] = -1e30f; l_arr[j] = 0.f; }

  for (int kb = 0; kb < 4; kb++){
    __syncthreads();                   // previous compute done with Ks/Vt
#if USE_TDM
    if (issuer)
      tdm_load_2d(&Ks[0], Kp + (size_t)kb*64*kHd, 64, 64, kHd, 4, 3);
    { // V: transpose into Vt (all 512 threads)
      int key = tid >> 3, c = tid & 7;
      v8h vv = *(const v8h*)(Vp + (size_t)(kb*64+key)*kHd + c*8);
#pragma unroll
      for (int i=0;i<8;i++) Vt[(c*8+i)*72 + key] = vv[i];
    }
    if (issuer) __builtin_amdgcn_s_wait_tensorcnt((short)0);
#else
    {
      int key = tid >> 3, c = tid & 7;
      *(v8h*)&Ks[key*72 + c*8] = *(const v8h*)(Kp + (size_t)(kb*64+key)*kHd + c*8);
      v8h vv = *(const v8h*)(Vp + (size_t)(kb*64+key)*kHd + c*8);
#pragma unroll
      for (int i=0;i<8;i++) Vt[(c*8+i)*72 + key] = vv[i];
    }
#endif
    __syncthreads();
    if (kb*4 > qt) continue;           // block entirely above causal diag

#pragma unroll 1
    for (int c2 = 0; c2 < 2; c2++){    // 32-key chunk
      int kt0 = kb*4 + c2*2;
      if (kt0 > qt) break;
      bool has1 = (kt0 + 1) <= qt;

      v8f s0 = vzero8();
#pragma unroll
      for (int t=0;t<2;t++){
        const _Float16* kr = &Ks[(c2*32 + l16)*72 + t*32 + (hf?16:0)];
        s0 = WMMA_F16(Aq[t], cat8(*(const v8h*)kr, *(const v8h*)(kr+8)), s0);
      }
      v8f s1 = vzero8();
      if (has1){
#pragma unroll
        for (int t=0;t<2;t++){
          const _Float16* kr = &Ks[(c2*32 + 16 + l16)*72 + t*32 + (hf?16:0)];
          s1 = WMMA_F16(Aq[t], cat8(*(const v8h*)kr, *(const v8h*)(kr+8)), s1);
        }
      }
#pragma unroll
      for (int j=0;j<8;j++){ s0[j] *= 0.125f; s1[j] *= 0.125f; }

      if (kt0 == qt){                  // diagonal tile mask: N > M
#pragma unroll
        for (int j=0;j<8;j++) if (l16 > j + 8*hf) s0[j] = -1e30f;
      }
      if (!has1){
#pragma unroll
        for (int j=0;j<8;j++) s1[j] = -1e30f;
      } else if (kt0 + 1 == qt){
#pragma unroll
        for (int j=0;j<8;j++) if (l16 > j + 8*hf) s1[j] = -1e30f;
      }

      float mnew[8], esc[8];
#pragma unroll
      for (int j=0;j<8;j++){
        float rm = fmaxf(s0[j], s1[j]);
#pragma unroll
        for (int msk=1; msk<16; msk<<=1)
          rm = fmaxf(rm, __shfl_xor(rm, msk, 32));   // 16-lane row-group reduce
        mnew[j] = fmaxf(m_arr[j], rm);
        esc[j]  = __expf(m_arr[j] - mnew[j]);
        m_arr[j] = mnew[j];
      }
      v8f p0, p1;
#pragma unroll
      for (int j=0;j<8;j++){
        p0[j] = __expf(s0[j] - mnew[j]);
        p1[j] = __expf(s1[j] - mnew[j]);
        float t = p0[j] + p1[j];
#pragma unroll
        for (int msk=1; msk<16; msk<<=1)
          t += __shfl_xor(t, msk, 32);
        l_arr[j] = l_arr[j]*esc[j] + t;
      }
#pragma unroll
      for (int nt=0;nt<4;nt++)
#pragma unroll
        for (int j=0;j<8;j++) acc[nt][j] *= esc[j];

      // C-layout f32 -> A-layout f16 via per-wave LDS tile
#pragma unroll
      for (int j=0;j<8;j++){
        int pr = j + 8*hf;
        Ps[wave][pr*40 + l16]      = (_Float16)p0[j];
        Ps[wave][pr*40 + 16 + l16] = (_Float16)p1[j];
      }
      asm volatile("s_wait_dscnt 0x0" ::: "memory");
      v16h Apm = cat8(*(const v8h*)&Ps[wave][l16*40 + ab],
                      *(const v8h*)&Ps[wave][l16*40 + 8 + ab]);
#pragma unroll
      for (int nt=0;nt<4;nt++){
        const _Float16* vr = &Vt[(nt*16 + l16)*72 + c2*32 + (hf?16:0)];
        acc[nt] = WMMA_F16(Apm, cat8(*(const v8h*)vr, *(const v8h*)(vr+8)), acc[nt]);
      }
    }
  }

  float linv[8];
#pragma unroll
  for (int j=0;j<8;j++) linv[j] = 1.f / l_arr[j];
#pragma unroll
  for (int nt=0;nt<4;nt++)
#pragma unroll
  for (int j=0;j<8;j++){
    size_t gr = (size_t)b*kS + (size_t)wdw*kW + qt*16 + j + 8*hf;
    a16[gr*kD + h*kHd + nt*16 + l16] = (_Float16)(acc[nt][j] * linv[j]);
  }
}

// ---------------------------------------------------------------------------
extern "C" void kernel_launch(void* const* d_in, const int* in_sizes, int n_in,
                              void* d_out, int out_size, void* d_ws, size_t ws_size,
                              hipStream_t stream) {
  (void)in_sizes; (void)n_in; (void)out_size; (void)ws_size;
  const float* x  = (const float*)d_in[0];
  const float* wq = (const float*)d_in[1];
  const float* wk = (const float*)d_in[2];
  const float* wv = (const float*)d_in[3];
  const float* wo = (const float*)d_in[4];
  const float* qg = (const float*)d_in[5];
  float* out = (float*)d_out;

  const size_t ME = (size_t)kB * kS * kD;          // 16,777,216 (x / a16 elems)
  const size_t WE = (size_t)kD * kD;               // 1,048,576 per weight
  _Float16* x16 = (_Float16*)d_ws;
  _Float16* w16 = x16 + ME;                        // [4][1024*1024]
  _Float16* q16 = w16 + 4*WE;
  _Float16* k16 = q16 + ME;
  _Float16* v16 = k16 + ME;
  _Float16* a16 = v16 + ME;

  // 0) one-time f32 -> f16 conversions (x + 4 weights)
  cvt_kernel<<<dim3(ME/2048), 256, 0, stream>>>(x,  x16);
  cvt_kernel<<<dim3(WE/2048), 256, 0, stream>>>(wq, w16 + 0*WE);
  cvt_kernel<<<dim3(WE/2048), 256, 0, stream>>>(wk, w16 + 1*WE);
  cvt_kernel<<<dim3(WE/2048), 256, 0, stream>>>(wv, w16 + 2*WE);
  cvt_kernel<<<dim3(WE/2048), 256, 0, stream>>>(wo, w16 + 3*WE);
  // 1) QKV projections (3 GEMMs via grid.z), TDM-staged, -> f16 [B][H][P][64]
  gemm_kernel<0><<<dim3(8, 128, 3), 256, 0, stream>>>(
      x16, w16, q16, k16, v16, nullptr);
  // 2) RMS norm + RoPE (+gain) in place on Q and K
  rmsrope_kernel<<<dim3((kB*kH*kS)/256, 2), 256, 0, stream>>>(q16, k16, qg);
  // 3) Windowed causal attention -> f16 [B*S][1024]
  attn_kernel<<<dim3(kH, kS/kW, kB), 512, 0, stream>>>(q16, k16, v16, a16);
  // 4) Output projection -> f32 d_out
  gemm_kernel<1><<<dim3(8, 128, 1), 256, 0, stream>>>(
      a16, w16 + 3*WE, nullptr, nullptr, nullptr, out);
}
